// EmployeeGCNEncoder_43233140802156
// MI455X (gfx1250) — matrix operand order repaired and checked by reading, not compiled, not run on hardware
//
#include <hip/hip_runtime.h>

// ---------------- problem constants (from reference) ----------------
#define N_NODES 100000
#define N_EDGES 1600000
#define DIM_IN  256
#define DIM_H   128

typedef __bf16 v16bf __attribute__((ext_vector_type(16)));
typedef float  v8f   __attribute__((ext_vector_type(8)));

union Frag { v16bf v; unsigned u[8]; };

// fp32 -> bf16 round-to-nearest-even, packed pair / single
__device__ __forceinline__ unsigned bf16pair(float lo, float hi) {
  unsigned a = __float_as_uint(lo), b = __float_as_uint(hi);
  a += 0x7FFFu + ((a >> 16) & 1u);
  b += 0x7FFFu + ((b >> 16) & 1u);
  return (a >> 16) | (b & 0xFFFF0000u);
}
__device__ __forceinline__ unsigned short bf16one(float x) {
  unsigned a = __float_as_uint(x);
  a += 0x7FFFu + ((a >> 16) & 1u);
  return (unsigned short)(a >> 16);
}

// ---------------- WMMA GEMM: out[n x 128] = act(A[n x K]) @ W[K x 128] ------
// 1 block = 128 rows x 128 cols, 8 waves; each wave computes 16 rows x 128 cols.
// W staged in LDS transposed (pairs along K contiguous), fp32->bf16 on the fly.
template <int K, bool RELU>
__global__ __launch_bounds__(256) void gcn_gemm_wmma(
    const float* __restrict__ A, const float* __restrict__ W,
    float* __restrict__ out, int nrows) {
  __shared__ unsigned short ldsW[DIM_H * 130];  // col-major chunk, pad 128->130

  const int tid  = threadIdx.x;
  const int lane = tid & 31;
  const int wave = tid >> 5;
  const int half = lane >> 4;   // lane group (0: lanes 0-15, 1: lanes 16-31)
  const int l16  = lane & 15;
  const int rowBase = blockIdx.x * 128 + wave * 16;

  v8f acc[8] = {};

  for (int k0 = 0; k0 < K; k0 += 128) {
    __syncthreads();
    // stage W[k0..k0+128) x 128 cols into LDS, transposed + bf16
    for (int idx = tid; idx < 128 * DIM_H; idx += 256) {
      int c = idx & 127, k = idx >> 7;                       // coalesced global
      ldsW[c * 130 + k] = bf16one(W[(size_t)(k0 + k) * DIM_H + c]);
    }
    __syncthreads();

    #pragma unroll
    for (int ks = 0; ks < 128; ks += 32) {
      // ---- A fragment: 16x32 bf16 (ISA layout: lanes 0-15 K 0-7/16-23,
      //      lanes 16-31 K 8-15/24-31, paired in VGPRs) ----
      Frag a;
      int arow = rowBase + l16;
      if (arow >= nrows) arow = nrows - 1;                   // clamp, EXEC all-1
      const float* ap = A + (size_t)arow * K + (k0 + ks) + half * 8;
      #pragma unroll
      for (int j = 0; j < 8; ++j) {
        int kk = ((j & 3) << 1) + ((j >> 2) << 4);
        float2 f = *(const float2*)(ap + kk);
        if (RELU) { f.x = fmaxf(f.x, 0.0f); f.y = fmaxf(f.y, 0.0f); }
        a.u[j] = bf16pair(f.x, f.y);
      }

      // ---- 8 column tiles: B frag from LDS (K-pairs contiguous) + WMMA ----
      #pragma unroll
      for (int ct = 0; ct < 8; ++ct) {
        Frag b;
        const unsigned short* wp =
            &ldsW[(ct * 16 + l16) * 130 + ks + half * 8];
        #pragma unroll
        for (int j = 0; j < 8; ++j)
          b.u[j] = *(const unsigned*)(wp + ((j & 3) << 1) + ((j >> 2) << 4));           // even -> aligned
        acc[ct] = __builtin_amdgcn_wmma_f32_16x16x32_bf16(
            false, a.v, false, b.v, (short)0, acc[ct], false, false);
      }
    }
  }

  // ---- store: C/D layout VGPR r -> M = half*8 + r, N = l16 ----
  #pragma unroll
  for (int ct = 0; ct < 8; ++ct) {
    #pragma unroll
    for (int r = 0; r < 8; ++r) {
      int row = rowBase + half * 8 + r;
      if (row < nrows)
        out[(size_t)row * DIM_H + ct * 16 + l16] = acc[ct][r];
    }
  }
}

// ---------------- degree / norm ----------------
__global__ void k_fill1(float* p, int n) {
  int i = blockIdx.x * 256 + threadIdx.x;
  if (i < n) p[i] = 1.0f;                      // self-loop contributes 1
}
__global__ void k_deg(const long long* __restrict__ dst, float* deg, int e) {
  int i = blockIdx.x * 256 + threadIdx.x;
  if (i < e) unsafeAtomicAdd(&deg[dst[i]], 1.0f);
}
__global__ void k_rsqrt(float* p, int n) {
  int i = blockIdx.x * 256 + threadIdx.x;
  if (i < n) p[i] = rsqrtf(p[i]);              // deg >= 1 always
}

// ---------------- self-loop + bias init: out = h*dinv^2 + b ----------------
__global__ void k_init_out(const float* __restrict__ h,
                           const float* __restrict__ dinv,
                           const float* __restrict__ bias,
                           float* __restrict__ out, int total) {
  int gid = blockIdx.x * 256 + threadIdx.x;
  if (gid >= total) return;
  int i = gid >> 7, c = gid & 127;
  float di = dinv[i];
  out[gid] = h[gid] * di * di + bias[c];
}

// ---------------- edge scatter-add (L2-resident, fp32 HW atomics) ----------
__global__ void k_edge(const long long* __restrict__ src,
                       const long long* __restrict__ dst,
                       const float* __restrict__ dinv,
                       const float* __restrict__ h,
                       float* __restrict__ out, int e) {
  int gid = blockIdx.x * 256 + threadIdx.x;
  if (gid >= e * 4) return;
  int ed = gid >> 2, q = gid & 3;              // 4 threads per edge, 32 cols ea
  long long s = src[ed], d = dst[ed];
  float nm = dinv[s] * dinv[d];
  const float4* hp = (const float4*)(h + (size_t)s * DIM_H + q * 32);
  float* op = out + (size_t)d * DIM_H + q * 32;
  #pragma unroll
  for (int j = 0; j < 8; ++j) {
    float4 v = hp[j];
    unsafeAtomicAdd(op + 4 * j + 0, v.x * nm);
    unsafeAtomicAdd(op + 4 * j + 1, v.y * nm);
    unsafeAtomicAdd(op + 4 * j + 2, v.z * nm);
    unsafeAtomicAdd(op + 4 * j + 3, v.w * nm);
  }
}

extern "C" void kernel_launch(void* const* d_in, const int* in_sizes, int n_in,
                              void* d_out, int out_size, void* d_ws, size_t ws_size,
                              hipStream_t stream) {
  (void)in_sizes; (void)n_in; (void)out_size; (void)ws_size;
  const float*     x   = (const float*)d_in[0];
  const long long* ei  = (const long long*)d_in[1];   // int64 per reference
  const float*     W1  = (const float*)d_in[2];
  const float*     b1  = (const float*)d_in[3];
  const float*     W2  = (const float*)d_in[4];
  const float*     b2  = (const float*)d_in[5];
  const float*     W3  = (const float*)d_in[6];
  const float*     b3  = (const float*)d_in[7];
  const long long* src = ei;
  const long long* dst = ei + N_EDGES;

  // workspace: dinv[N] | bufH[N*128] | bufO[N*128]  (~103 MB)
  float* dinv = (float*)d_ws;
  size_t off  = ((size_t)N_NODES * 4 + 511) & ~(size_t)511;
  float* bufH = (float*)((char*)d_ws + off);
  float* bufO = bufH + (size_t)N_NODES * DIM_H;
  float* fout = (float*)d_out;

  dim3 blk(256);
  int gN  = (N_NODES + 255) / 256;
  int gE  = (N_EDGES + 255) / 256;
  int gNH = (N_NODES * DIM_H + 255) / 256;
  int gE4 = (N_EDGES * 4 + 255) / 256;
  int gM  = (N_NODES + 127) / 128;

  // degree -> dinv
  k_fill1 <<<gN, blk, 0, stream>>>(dinv, N_NODES);
  k_deg   <<<gE, blk, 0, stream>>>(dst, dinv, N_EDGES);
  k_rsqrt <<<gN, blk, 0, stream>>>(dinv, N_NODES);

  // layer 1 (K=256, no input relu)
  gcn_gemm_wmma<DIM_IN, false><<<gM, blk, 0, stream>>>(x, W1, bufH, N_NODES);
  k_init_out<<<gNH, blk, 0, stream>>>(bufH, dinv, b1, bufO, N_NODES * DIM_H);
  k_edge    <<<gE4, blk, 0, stream>>>(src, dst, dinv, bufH, bufO, N_EDGES);

  // layer 2 (relu fused into A-load)
  gcn_gemm_wmma<DIM_H, true><<<gM, blk, 0, stream>>>(bufO, W2, bufH, N_NODES);
  k_init_out<<<gNH, blk, 0, stream>>>(bufH, dinv, b2, bufO, N_NODES * DIM_H);
  k_edge    <<<gE4, blk, 0, stream>>>(src, dst, dinv, bufH, bufO, N_EDGES);

  // layer 3 (relu fused; aggregation lands directly in d_out)
  gcn_gemm_wmma<DIM_H, true><<<gM, blk, 0, stream>>>(bufO, W3, bufH, N_NODES);
  k_init_out<<<gNH, blk, 0, stream>>>(bufH, dinv, b3, fout, N_NODES * DIM_H);
  k_edge    <<<gE4, blk, 0, stream>>>(src, dst, dinv, bufH, fout, N_EDGES);
}
